// FourthOrderPooling_59837484368487
// MI455X (gfx1250) — compile-verified, matrix operand order checked by reading, not run on hardware
//
#include <hip/hip_runtime.h>

typedef __attribute__((ext_vector_type(16))) __bf16 v16bf;
typedef __attribute__((ext_vector_type(8)))  float  v8f;
typedef __attribute__((ext_vector_type(16))) unsigned short v16u;
typedef __attribute__((ext_vector_type(8)))  unsigned short v8u;

#define B_SZ    64
#define N_TOK   4096
#define D_DIM   128
#define OUTD    512
#define COVP    384
#define LDS_K   40   // padded k-stride (halves): 16B-aligned frag reads, conflict-spread

// native f32->bf16 (backend emits v_cvt_(pk_)bf16_f32), packed pair
__device__ __forceinline__ unsigned int pk2(float a, float b) {
  const unsigned short ua = __builtin_bit_cast(unsigned short, (__bf16)a);
  const unsigned short ub = __builtin_bit_cast(unsigned short, (__bf16)b);
  return (unsigned int)ua | ((unsigned int)ub << 16);
}

// ---------------------------------------------------------------------------
// K1: single-pass raw moments -> mean, sq_mean, kurtosis (kurt written into
//     combined[:, 0:128]).  One block per batch, 256 threads (8 waves).
// ---------------------------------------------------------------------------
__global__ void __launch_bounds__(256) moments_kernel(
    const float* __restrict__ x, float* __restrict__ mean,
    float* __restrict__ sqm, float* __restrict__ combined) {
  const int b   = blockIdx.x;
  const int tid = threadIdx.x;
  const int d4  = (tid & 31) * 4;   // fixed 4 consecutive d per thread
  const int g   = tid >> 5;         // wave id: n-stride group
  float s1[4] = {0,0,0,0}, s2[4] = {0,0,0,0}, s3[4] = {0,0,0,0}, s4[4] = {0,0,0,0};
  const float* xb = x + (size_t)b * N_TOK * D_DIM;
  for (int n = g; n < N_TOK; n += 8) {
    const float4 xv = *(const float4*)(xb + (size_t)n * D_DIM + d4);
    const float v[4] = {xv.x, xv.y, xv.z, xv.w};
#pragma unroll
    for (int j = 0; j < 4; ++j) {
      const float t = v[j], t2 = t * t;
      s1[j] += t; s2[j] += t2; s3[j] += t2 * t; s4[j] += t2 * t2;
    }
  }
  __shared__ float red[4][8][D_DIM];
#pragma unroll
  for (int j = 0; j < 4; ++j) {
    red[0][g][d4 + j] = s1[j];
    red[1][g][d4 + j] = s2[j];
    red[2][g][d4 + j] = s3[j];
    red[3][g][d4 + j] = s4[j];
  }
  __syncthreads();
  if (tid < D_DIM) {
    float m1 = 0.f, m2 = 0.f, m3 = 0.f, m4 = 0.f;
#pragma unroll
    for (int w = 0; w < 8; ++w) {
      m1 += red[0][w][tid]; m2 += red[1][w][tid];
      m3 += red[2][w][tid]; m4 += red[3][w][tid];
    }
    const float inv = 1.0f / (float)N_TOK;
    m1 *= inv; m2 *= inv; m3 *= inv; m4 *= inv;
    const float mu   = m1;
    const float var  = m2 - mu * mu;                                    // sq_mean
    const float kurt = m4 - 4.f*mu*m3 + 6.f*mu*mu*m2 - 3.f*mu*mu*mu*mu; // E[(x-mu)^4]
    mean[b * D_DIM + tid] = mu;
    sqm [b * D_DIM + tid] = var;
    combined[(size_t)b * OUTD + tid] = kurt;
  }
}

// ---------------------------------------------------------------------------
// K2: cov_sq[b] = sqc^T sqc / (N-1); sqc computed on the fly; bf16 WMMA.
//     Double-buffered LDS tile of sqc transposed: [d=128][k=32] bf16,
//     k-stride 40 halves.  One barrier per K-iteration; staging for tile i+1
//     overlaps fragment loads / WMMAs of tile i.
//     A frag (row tile d0): A[m][k] = sqc[k][d0+m] -> k-contiguous in LDS.
//     B frag (col tile e0): B[k][n] = sqc[k][e0+n] -> same loader.
// ---------------------------------------------------------------------------
__device__ __forceinline__ v16bf lds_frag(const unsigned short* lds, int row, int khalf) {
  const unsigned short* p = lds + row * LDS_K + khalf;
  const v8u lo = *(const v8u*)(p);        // K = khalf .. khalf+7
  const v8u hi = *(const v8u*)(p + 16);   // K = 16+khalf .. 16+khalf+7
  const v16u r = __builtin_shufflevector(lo, hi, 0,1,2,3,4,5,6,7,8,9,10,11,12,13,14,15);
  return __builtin_bit_cast(v16bf, r);
}

__global__ void __launch_bounds__(256) cov_wmma_kernel(
    const float* __restrict__ x, const float* __restrict__ mean,
    const float* __restrict__ sqm, float* __restrict__ cov) {
  const int b    = blockIdx.x;
  const int tid  = threadIdx.x;
  const int wave = tid >> 5;
  const int lane = tid & 31;

  __shared__ alignas(16) unsigned short sq_t[2][D_DIM * LDS_K]; // bf16 bits, [d][k]

  // staging mapping: thread owns feature row ds, packs 4 consecutive k as b64
  const int   ds_  = lane + (wave & 3) * 32;   // 0..127
  const int   ns0  = (wave >> 2) * 4;          // n-group base (0 or 4)
  const float mu   = mean[b * D_DIM + ds_];
  const float var  = sqm [b * D_DIM + ds_];

  const int dA    = (wave & 3) * 32;   // 2 row tiles per wave
  const int dB    = (wave >> 2) * 64;  // 4 col tiles per wave
  const int mrow  = lane & 15;
  const int khalf = (lane >> 4) * 8;
  v8f acc[2][4] = {};

  const float* xb = x + (size_t)b * N_TOK * D_DIM;

  auto stage = [&](int k0, int bufi) {
#pragma unroll
    for (int it = 0; it < 4; ++it) {
      const int n = ns0 + it * 8;            // 4-aligned -> 8B-aligned b64 store
      float c[4];
#pragma unroll
      for (int j = 0; j < 4; ++j) {          // wave-coalesced 128B row segments
        const float t = xb[(size_t)(k0 + n + j) * D_DIM + ds_] - mu;
        c[j] = t * t - var;
      }
      const unsigned long long pk =
          (unsigned long long)pk2(c[0], c[1]) |
          ((unsigned long long)pk2(c[2], c[3]) << 32);
      *(unsigned long long*)&sq_t[bufi][ds_ * LDS_K + n] = pk;
    }
  };

  stage(0, 0);
  __syncthreads();

  const int NIT = N_TOK / 32;
  for (int i = 0; i < NIT; ++i) {
    const int cur = i & 1;
    const unsigned short* buf = sq_t[cur];

    const v16bf afr0 = lds_frag(buf, dA + mrow,      khalf);
    const v16bf afr1 = lds_frag(buf, dA + 16 + mrow, khalf);
    v16bf bfr[4];
#pragma unroll
    for (int c = 0; c < 4; ++c) bfr[c] = lds_frag(buf, dB + c * 16 + mrow, khalf);

    if (i + 1 < NIT) {
      stage((i + 1) * 32, cur ^ 1);          // overlap with WMMAs below
      __builtin_prefetch(xb + (size_t)((i + 2) * 32 + ns0) * D_DIM + ds_, 0, 0);
    }

#pragma unroll
    for (int c = 0; c < 4; ++c) {
      acc[0][c] = __builtin_amdgcn_wmma_f32_16x16x32_bf16(false, afr0, false, bfr[c],
                                                          (short)0, acc[0][c], false, false);
      acc[1][c] = __builtin_amdgcn_wmma_f32_16x16x32_bf16(false, afr1, false, bfr[c],
                                                          (short)0, acc[1][c], false, false);
    }
    __syncthreads();   // single barrier per iteration (double-buffered)
  }

  // epilogue: C/D layout -> lanes 0-15: M=r, N=lane; lanes 16-31: M=r+8, N=lane-16
  const float scale = 1.0f / (float)(N_TOK - 1);
  const int   mhi   = (lane >> 4) * 8;
  const int   ncol  = lane & 15;
  float* cb = cov + (size_t)b * D_DIM * D_DIM;
#pragma unroll
  for (int rt = 0; rt < 2; ++rt)
#pragma unroll
    for (int ct = 0; ct < 4; ++ct)
#pragma unroll
      for (int r = 0; r < 8; ++r) {
        const int row = dA + rt * 16 + r + mhi;
        const int col = dB + ct * 16 + ncol;
        cb[(size_t)row * D_DIM + col] = acc[rt][ct][r] * scale;
      }
}

// ---------------------------------------------------------------------------
// K3/K4: C(64 x N) = A(64 x K) @ W(N x K)^T + bias,  fp32 in/out, bf16 WMMA.
//   grid.x = N/16; 8 waves split K and sum their partials through LDS.
//   Fragments built directly from global (A and W are both K-contiguous).
// ---------------------------------------------------------------------------
__device__ __forceinline__ v16bf gfrag(const float* __restrict__ p) {
  v16bf r;
#pragma unroll
  for (int i = 0; i < 8; ++i) r[i] = (__bf16)p[i];
#pragma unroll
  for (int i = 0; i < 8; ++i) r[8 + i] = (__bf16)p[16 + i];
  return r;
}

__global__ void __launch_bounds__(256) gemm64_wmma_bias(
    const float* __restrict__ A, int lda,
    const float* __restrict__ W, int ldw,
    const float* __restrict__ bias,
    float* __restrict__ C, int ldc, int K) {
  const int tid   = threadIdx.x;
  const int wave  = tid >> 5;
  const int lane  = tid & 31;
  const int nbase = blockIdx.x * 16;
  const int Kc    = K >> 3;            // K multiple of 256
  const int mrow  = lane & 15;
  const int khalf = (lane >> 4) * 8;

  v8f acc[4] = {};
  const float* wrow = W + (size_t)(nbase + mrow) * ldw + khalf;
  const int kend = (wave + 1) * Kc;
  for (int k0 = wave * Kc; k0 < kend; k0 += 32) {
    const v16bf bfr = gfrag(wrow + k0);
#pragma unroll
    for (int t = 0; t < 4; ++t) {
      const v16bf afr = gfrag(A + (size_t)(t * 16 + mrow) * lda + k0 + khalf);
      acc[t] = __builtin_amdgcn_wmma_f32_16x16x32_bf16(false, afr, false, bfr,
                                                       (short)0, acc[t], false, false);
    }
  }

  __shared__ float red[8][4][256];
#pragma unroll
  for (int t = 0; t < 4; ++t)
#pragma unroll
    for (int r = 0; r < 8; ++r)
      red[wave][t][r * 32 + lane] = acc[t][r];
  __syncthreads();

  const int r      = tid >> 5;
  const int l      = tid & 31;
  const int col    = nbase + (l & 15);
  const int rowoff = r + ((l >= 16) ? 8 : 0);
#pragma unroll
  for (int t = 0; t < 4; ++t) {
    float s = 0.f;
#pragma unroll
    for (int w = 0; w < 8; ++w) s += red[w][t][tid];
    C[(size_t)(t * 16 + rowoff) * ldc + col] = s + bias[col];
  }
}

// ---------------------------------------------------------------------------
extern "C" void kernel_launch(void* const* d_in, const int* in_sizes, int n_in,
                              void* d_out, int out_size, void* d_ws, size_t ws_size,
                              hipStream_t stream) {
  const float* x       = (const float*)d_in[0];
  const float* W_cov   = (const float*)d_in[1];
  const float* b_cov   = (const float*)d_in[2];
  const float* W_final = (const float*)d_in[3];
  const float* b_final = (const float*)d_in[4];
  float* out = (float*)d_out;

  // workspace layout (floats): mean 8K | sq_mean 8K | combined 32K | cov 1M
  float* ws       = (float*)d_ws;
  float* mean     = ws;                          // 64*128
  float* sqm      = ws + 8192;                   // 64*128
  float* combined = ws + 16384;                  // 64*512
  float* cov      = ws + 49152;                  // 64*128*128  (~4.2 MB total)

  moments_kernel <<<B_SZ, 256, 0, stream>>>(x, mean, sqm, combined);
  cov_wmma_kernel<<<B_SZ, 256, 0, stream>>>(x, mean, sqm, cov);
  // cov_proj = cov_flat @ W_cov^T + b_cov  -> combined[:, 128:512]
  gemm64_wmma_bias<<<COVP / 16, 256, 0, stream>>>(cov, D_DIM * D_DIM,
                                                  W_cov, D_DIM * D_DIM, b_cov,
                                                  combined + D_DIM, OUTD,
                                                  D_DIM * D_DIM);
  // out = combined @ W_final^T + b_final
  gemm64_wmma_bias<<<OUTD / 16, 256, 0, stream>>>(combined, OUTD,
                                                  W_final, OUTD, b_final,
                                                  out, OUTD, OUTD);
}
